// Mamba2Block_46282567581797
// MI455X (gfx1250) — matrix operand order repaired and checked by reading, not compiled, not run on hardware
//
#include <hip/hip_runtime.h>
#include <hip/hip_bf16.h>

// ---------------------------------------------------------------------------
// Mamba2 block forward for MI455X (gfx1250), wave32 + WMMA bf16.
//
// All matmuls run through v_wmma_f32_16x16x32_bf16 (fp32 accumulate).
// Fragment layouts follow CDNA5 ISA 7.12.2 (wave32):
//   A 16x32 bf16 : lanes 0-15 row=lane,   e0..7=K0..7,  e8..15=K16..23
//                  lanes 16-31 row=lane-16,e0..7=K8..15, e8..15=K24..31
//   B 32x16 bf16 : lanes 0-15 col=lane,   e0..7=K0..7,  e8..15=K16..23
//                  lanes 16-31 col=lane-16,e0..7=K8..15, e8..15=K24..31
//   C/D 16x16 f32: vgpr e, lanes 0-15 -> (m=e,   n=lane)
//                          lanes 16-31 -> (m=8+e, n=lane-16)
//
// GEMM tiles are staged with GLOBAL_LOAD_ASYNC_TO_LDS_B128 (ASYNCcnt) and
// double-buffered: next tile's async copies are issued before computing the
// current one; s_wait_asynccnt(8) + barrier gates the flip.
// ---------------------------------------------------------------------------

typedef __attribute__((ext_vector_type(16))) __bf16 v16bf;
typedef __attribute__((ext_vector_type(8)))  __bf16 v8bf;
typedef __attribute__((ext_vector_type(8)))  float  v8f;
typedef __attribute__((ext_vector_type(4)))  int    v4i;

#define LSEQ   4096
#define DMODEL 2048
#define DINNER 4096
#define NHEADS 64
#define DHEAD  64
#define DSTATE 64
#define NCHUNK 16
#define QCHUNK 256
#define NPR    8256   // N_HEADS * (1 + 2*64)
#define XRN    8192   // 2 * D_INNER

#if __has_builtin(__builtin_amdgcn_global_load_async_to_lds_b128) && \
    __has_builtin(__builtin_amdgcn_s_wait_asynccnt)
#define ASYNC_LDS 1
#endif

__device__ __forceinline__ v8f wmma_bf16(v16bf a, v16bf b, v8f c) {
  return __builtin_amdgcn_wmma_f32_16x16x32_bf16(false, a, false, b,
                                                 (short)0, c, false, false);
}

__device__ __forceinline__ v8f v8f_zero() {
  v8f z;
#pragma unroll
  for (int e = 0; e < 8; ++e) z[e] = 0.0f;
  return z;
}

#ifdef ASYNC_LDS
// 16-byte global -> LDS async copy (gfx1250 GLOBAL_LOAD_ASYNC_TO_LDS_B128).
// Toolchain prototype takes v4i* operands (probe: round-2 diagnostic).
__device__ __forceinline__ void async_cp16(const __bf16* g, __bf16* l) {
  __builtin_amdgcn_global_load_async_to_lds_b128((v4i*)g, (v4i*)l, 0, 0);
}
#endif

// Load a 16x32 A-fragment from an LDS tile (row-major, ld elems).
__device__ __forceinline__ v16bf load_a_frag(const __bf16* tile, int ld,
                                             int row0, int k0, int lane) {
  int r = lane & 15, hb = lane >> 4;
  const __bf16* p = tile + (row0 + r) * ld + k0 + hb * 8;
  union { v16bf v; v8bf h[2]; } u;
  u.h[0] = *(const v8bf*)(p);
  u.h[1] = *(const v8bf*)(p + 16);
  return u.v;
}

// Load a 32x16 B-fragment; "columns" of B are rows of the tile (tile is
// [N rows][K cols], row-major, ld elems) -> same k order as the A-fragment.
__device__ __forceinline__ v16bf load_b_frag(const __bf16* tile, int ld,
                                             int col0, int k0, int lane) {
  int c = lane & 15, hb = lane >> 4;
  const __bf16* p = tile + (col0 + c) * ld + k0 + hb * 8;
  union { v16bf v; v8bf h[2]; } u;
  u.h[0] = *(const v8bf*)(p);
  u.h[1] = *(const v8bf*)(p + 16);
  return u.v;
}

// ---------------------------------------------------------------------------
// fp32 -> bf16 conversion
// ---------------------------------------------------------------------------
__global__ void __launch_bounds__(256)
cvt_f32_bf16(const float* __restrict__ src, __bf16* __restrict__ dst, size_t n) {
  size_t i = (size_t)blockIdx.x * blockDim.x + threadIdx.x;
  size_t stride = (size_t)gridDim.x * blockDim.x;
  for (; i < n; i += stride) dst[i] = (__bf16)src[i];
}

// ---------------------------------------------------------------------------
// Generic GEMM: C[M,N] f32 = A[M,K]bf16 @ B[N,K]bf16^T
// 128x128 block tile, 64-wide K stages in LDS, 8 waves of 32x64.
// M, K multiples of 128/64; N guarded (stores guarded; async loads clamp row).
// ---------------------------------------------------------------------------
#define GLD 80   // LDS row pitch (160B row stride keeps 32B alignment)

__global__ void __launch_bounds__(256)
gemm_bf16_nt(const __bf16* __restrict__ A, const __bf16* __restrict__ B,
             float* __restrict__ C, int M, int N, int K) {
#ifdef ASYNC_LDS
  __shared__ __bf16 As[2][128 * GLD];
  __shared__ __bf16 Bs[2][128 * GLD];
#else
  __shared__ __bf16 As[1][128 * GLD];
  __shared__ __bf16 Bs[1][128 * GLD];
#endif

  const int m0 = blockIdx.y * 128;
  const int n0 = blockIdx.x * 128;
  const int tid = threadIdx.x, lane = tid & 31, w = tid >> 5;
  const int wm = (w >> 1) * 32;   // wave row offset (0..96)
  const int wn = (w & 1) * 64;    // wave col offset (0/64)

  v8f acc[2][4];
#pragma unroll
  for (int i = 0; i < 2; ++i)
#pragma unroll
    for (int j = 0; j < 4; ++j) acc[i][j] = v8f_zero();

  const int rowa = tid >> 3;        // 0..31
  const int ca   = (tid & 7) * 8;   // 0..56

  auto compute = [&](const __bf16* Asb, const __bf16* Bsb) {
#pragma unroll
    for (int kk = 0; kk < 64; kk += 32) {
      v16bf af0 = load_a_frag(Asb, GLD, wm,      kk, lane);
      v16bf af1 = load_a_frag(Asb, GLD, wm + 16, kk, lane);
      v16bf bfr[4];
#pragma unroll
      for (int j = 0; j < 4; ++j)
        bfr[j] = load_b_frag(Bsb, GLD, wn + j * 16, kk, lane);
#pragma unroll
      for (int j = 0; j < 4; ++j) {
        acc[0][j] = wmma_bf16(af0, bfr[j], acc[0][j]);
        acc[1][j] = wmma_bf16(af1, bfr[j], acc[1][j]);
      }
    }
  };

#ifdef ASYNC_LDS
  // 8 async b128 copies per thread per tile (uniform -> ASYNCcnt += 8 / wave).
  auto stage = [&](int buf, int k0) {
#pragma unroll
    for (int rr = 0; rr < 128; rr += 32)
      async_cp16(A + (size_t)(m0 + rowa + rr) * K + k0 + ca,
                 &As[buf][(rowa + rr) * GLD + ca]);
#pragma unroll
    for (int rr = 0; rr < 128; rr += 32) {
      int n = n0 + rowa + rr;
      if (n > N - 1) n = N - 1;  // clamp: keeps count uniform; cols >= N unstored
      async_cp16(B + (size_t)n * K + k0 + ca,
                 &Bs[buf][(rowa + rr) * GLD + ca]);
    }
  };

  stage(0, 0);
  int buf = 0;
  for (int k0 = 0; k0 < K; k0 += 64) {
    const bool more = (k0 + 64) < K;
    if (more) stage(buf ^ 1, k0 + 64);   // prefetch next tile into other buffer
    if (more) __builtin_amdgcn_s_wait_asynccnt(8);   // current tile resident
    else      __builtin_amdgcn_s_wait_asynccnt(0);
    __syncthreads();                      // all waves' copies visible
    compute(&As[buf][0], &Bs[buf][0]);
    __syncthreads();                      // done reading before next overwrite
    buf ^= 1;
  }
#else
  for (int k0 = 0; k0 < K; k0 += 64) {
#pragma unroll
    for (int rr = 0; rr < 128; rr += 32) {
      v8bf va = *(const v8bf*)(A + (size_t)(m0 + rowa + rr) * K + k0 + ca);
      *(v8bf*)(&As[0][(rowa + rr) * GLD + ca]) = va;
    }
#pragma unroll
    for (int rr = 0; rr < 128; rr += 32) {
      int n = n0 + rowa + rr;
      v8bf vb;
      if (n < N) {
        vb = *(const v8bf*)(B + (size_t)n * K + k0 + ca);
      } else {
#pragma unroll
        for (int q = 0; q < 8; ++q) vb[q] = (__bf16)0.0f;
      }
      *(v8bf*)(&Bs[0][(rowa + rr) * GLD + ca]) = vb;
    }
    if (k0 + 64 < K) {
      __builtin_prefetch(A + (size_t)(m0 + rowa) * K + k0 + 64 + ca, 0, 1);
      __builtin_prefetch(B + (size_t)(n0 + rowa) * K + k0 + 64 + ca, 0, 1);
    }
    __syncthreads();
    compute(&As[0][0], &Bs[0][0]);
    __syncthreads();
  }
#endif

  const int nl = lane & 15, hb = lane >> 4;
#pragma unroll
  for (int i = 0; i < 2; ++i)
#pragma unroll
    for (int j = 0; j < 4; ++j)
#pragma unroll
      for (int e = 0; e < 8; ++e) {
        int m = m0 + wm + i * 16 + hb * 8 + e;
        int n = n0 + wn + j * 16 + nl;
        if (n < N) C[(size_t)m * N + n] = acc[i][j][e];
      }
}

// ---------------------------------------------------------------------------
// Prep: per (chunk, head): softplus -> la, cumsum(cs), and bf16 operand
// tensors in WMMA-friendly layouts.
//   Bb/Cb/Ceb : [head][l][64] row-major (Ceb = exp(cs)*C)
//   BdT       : [head][chunk][64 d][256 j] = exp(cs_last - cs[j]) * B[j,d]
//   XcT       : [head][chunk][64 h][256 j] = X[j,h]
// ---------------------------------------------------------------------------
__global__ void __launch_bounds__(256)
prep_kernel(const float* __restrict__ params, const float* __restrict__ xr,
            float* __restrict__ cs_buf, float* __restrict__ Achunk,
            __bf16* __restrict__ Bb, __bf16* __restrict__ Cb,
            __bf16* __restrict__ Ceb, __bf16* __restrict__ BdT,
            __bf16* __restrict__ XcT) {
  const int c = blockIdx.x, h = blockIdx.y, t = threadIdx.x;
  __shared__ float cs[QCHUNK];
  const int tok = c * QCHUNK + t;
  const float* prow = params + (size_t)tok * NPR + h * 129;

  float raw = prow[0];
  float sp = raw > 20.0f ? raw : __logf(1.0f + __expf(raw));
  cs[t] = -sp;  // la = log(exp(-softplus)) = -softplus
  __syncthreads();
  // Hillis-Steele inclusive scan over the chunk
  for (int off = 1; off < QCHUNK; off <<= 1) {
    float v = (t >= off) ? cs[t - off] : 0.0f;
    __syncthreads();
    cs[t] += v;
    __syncthreads();
  }

  float mycs = cs[t];
  float cs_last = cs[QCHUNK - 1];
  cs_buf[(size_t)h * LSEQ + tok] = mycs;
  if (t == QCHUNK - 1) Achunk[h * NCHUNK + c] = __expf(cs_last);

  float ecs = __expf(mycs);             // <= 1 (cs <= 0)
  float dec = __expf(cs_last - mycs);   // <= 1

  size_t rb = ((size_t)h * LSEQ + tok) * 64;
  const float* Bp = prow + 1;
  const float* Cp = prow + 1 + DSTATE;
  const float* Xp = xr + (size_t)tok * XRN + h * DHEAD;
  size_t tbase = (((size_t)h * NCHUNK + c) * 64) * 256 + t;
#pragma unroll 8
  for (int d = 0; d < 64; ++d) {
    float bv = Bp[d], cv = Cp[d], xv = Xp[d];
    Bb[rb + d]  = (__bf16)bv;
    Cb[rb + d]  = (__bf16)cv;
    Ceb[rb + d] = (__bf16)(ecs * cv);
    BdT[tbase + (size_t)d * 256] = (__bf16)(dec * bv);
    XcT[tbase + (size_t)d * 256] = (__bf16)xv;
  }
}

// ---------------------------------------------------------------------------
// Diag: per (row-half, chunk, head): scores = (Cc Bc^T) .* L, then
// y_diag = scores @ Xc, processed in 64-wide j blocks with LDS P staging.
// 8 waves x 16 rows = 128 rows per block.
// ---------------------------------------------------------------------------
__global__ void __launch_bounds__(256)
diag_kernel(const __bf16* __restrict__ Cb, const __bf16* __restrict__ Bb,
            const __bf16* __restrict__ XcT, const float* __restrict__ cs_buf,
            float* __restrict__ ydiag) {
  const int half = blockIdx.x & 1, c = blockIdx.x >> 1, h = blockIdx.y;
  const int tid = threadIdx.x, lane = tid & 31, w = tid >> 5;
  const int i0 = half * 128;

  __shared__ __bf16 CcW[128 * 80];
  __shared__ __bf16 BbJ[64 * 80];
  __shared__ __bf16 XTJ[64 * 80];
  __shared__ __bf16 Pst[8][16 * 80];
  __shared__ float  cs_sh[QCHUNK];

  cs_sh[tid] = cs_buf[(size_t)h * LSEQ + c * QCHUNK + tid];
  {
    int r = tid >> 1, d0 = (tid & 1) * 32;
    const __bf16* src = Cb + ((size_t)h * LSEQ + c * QCHUNK + i0 + r) * 64 + d0;
    *(v16bf*)(CcW + r * 80 + d0)      = *(const v16bf*)(src);
    *(v16bf*)(CcW + r * 80 + d0 + 16) = *(const v16bf*)(src + 16);
  }

  v8f yacc[4];
#pragma unroll
  for (int j = 0; j < 4; ++j) yacc[j] = v8f_zero();
  const int nl = lane & 15, hb = lane >> 4;

  for (int jb = 0; jb < 4; ++jb) {
    __syncthreads();
    {
      int r = tid >> 2, q0 = (tid & 3) * 16;
      *(v16bf*)(BbJ + r * 80 + q0) =
          *(const v16bf*)(Bb + ((size_t)h * LSEQ + c * QCHUNK + jb * 64 + r) * 64 + q0);
      *(v16bf*)(XTJ + r * 80 + q0) =
          *(const v16bf*)(XcT + (((size_t)h * NCHUNK + c) * 64 + r) * 256 + jb * 64 + q0);
    }
    __syncthreads();

    // P(16x64) = Cc(rows) @ BbJ^T over K=64
    v8f pacc[4];
#pragma unroll
    for (int j = 0; j < 4; ++j) pacc[j] = v8f_zero();
#pragma unroll
    for (int kk = 0; kk < 64; kk += 32) {
      v16bf a = load_a_frag(CcW, 80, w * 16, kk, lane);
      v16bf bfr[4];
#pragma unroll
      for (int fn = 0; fn < 4; ++fn)
        bfr[fn] = load_b_frag(BbJ, 80, fn * 16, kk, lane);
#pragma unroll
      for (int fn = 0; fn < 4; ++fn)
        pacc[fn] = wmma_bf16(a, bfr[fn], pacc[fn]);
    }
    // apply causal decay mask and stage P (bf16) per-wave
#pragma unroll
    for (int fn = 0; fn < 4; ++fn)
#pragma unroll
      for (int e = 0; e < 8; ++e) {
        int ic = i0 + w * 16 + hb * 8 + e;  // chunk-local row
        int jc = jb * 64 + fn * 16 + nl;    // chunk-local col
        float v = pacc[fn][e];
        v = (jc <= ic) ? v * __expf(cs_sh[ic] - cs_sh[jc]) : 0.0f;
        Pst[w][(hb * 8 + e) * 80 + fn * 16 + nl] = (__bf16)v;
      }
    // yacc += P @ Xc (B operand from XTJ = X^T block)
#pragma unroll
    for (int kk = 0; kk < 64; kk += 32) {
      v16bf a = load_a_frag(&Pst[w][0], 80, 0, kk, lane);
      v16bf bfr[4];
#pragma unroll
      for (int fh = 0; fh < 4; ++fh)
        bfr[fh] = load_b_frag(XTJ, 80, fh * 16, kk, lane);
#pragma unroll
      for (int fh = 0; fh < 4; ++fh)
        yacc[fh] = wmma_bf16(a, bfr[fh], yacc[fh]);
    }
  }

#pragma unroll
  for (int fh = 0; fh < 4; ++fh)
#pragma unroll
    for (int e = 0; e < 8; ++e) {
      int ic = i0 + w * 16 + hb * 8 + e;
      int hh = fh * 16 + nl;
      ydiag[((size_t)h * LSEQ + c * QCHUNK + ic) * 64 + hh] = yacc[fh][e];
    }
}

// ---------------------------------------------------------------------------
// States: per (chunk, head): states[d][h] = sum_j decay[j]*B[j,d]*X[j,h]
// = BdT(64x256) @ XcT^T ... both operands prepared transposed by prep.
// ---------------------------------------------------------------------------
__global__ void __launch_bounds__(256)
states_kernel(const __bf16* __restrict__ BdT, const __bf16* __restrict__ XcT,
              float* __restrict__ states) {
  const int c = blockIdx.x, h = blockIdx.y;
  const int tid = threadIdx.x, lane = tid & 31, w = tid >> 5;
  __shared__ __bf16 Bt[64 * 144];
  __shared__ __bf16 Xt[64 * 144];

  const int d0 = (w >> 1) * 16;
  const int h0 = (w & 1) * 32;
  v8f acc[2];
  acc[0] = v8f_zero(); acc[1] = v8f_zero();

  for (int kh = 0; kh < 2; ++kh) {
    __syncthreads();
    {
      int r = tid >> 2, q0 = (tid & 3) * 32;
      size_t srcb = (((size_t)h * NCHUNK + c) * 64 + r) * 256 + kh * 128 + q0;
      *(v16bf*)(Bt + r * 144 + q0)      = *(const v16bf*)(BdT + srcb);
      *(v16bf*)(Bt + r * 144 + q0 + 16) = *(const v16bf*)(BdT + srcb + 16);
      *(v16bf*)(Xt + r * 144 + q0)      = *(const v16bf*)(XcT + srcb);
      *(v16bf*)(Xt + r * 144 + q0 + 16) = *(const v16bf*)(XcT + srcb + 16);
    }
    __syncthreads();
#pragma unroll
    for (int kk = 0; kk < 128; kk += 32) {
      v16bf a = load_a_frag(Bt, 144, d0, kk, lane);
      v16bf bfr[2];
#pragma unroll
      for (int u = 0; u < 2; ++u)
        bfr[u] = load_b_frag(Xt, 144, h0 + u * 16, kk, lane);
#pragma unroll
      for (int u = 0; u < 2; ++u)
        acc[u] = wmma_bf16(a, bfr[u], acc[u]);
    }
  }

  const int nl = lane & 15, hb = lane >> 4;
#pragma unroll
  for (int u = 0; u < 2; ++u)
#pragma unroll
    for (int e = 0; e < 8; ++e) {
      int d = d0 + hb * 8 + e;
      int hh = h0 + u * 16 + nl;
      states[(((size_t)h * NCHUNK + c) * 64 + d) * 64 + hh] = acc[u][e];
    }
}

// ---------------------------------------------------------------------------
// Scan: sequential over 16 chunks per head; S_prev emitted transposed (bf16)
// as [head][chunk][h][d] so it can be the WMMA B operand of y_inter.
// ---------------------------------------------------------------------------
__global__ void __launch_bounds__(256)
scan_kernel(const float* __restrict__ states, const float* __restrict__ Achunk,
            __bf16* __restrict__ SprevT) {
  const int h = blockIdx.x, tid = threadIdx.x;
  const int d = tid >> 2;
  const int hh0 = (tid & 3) * 16;
  float S[16];
#pragma unroll
  for (int k = 0; k < 16; ++k) S[k] = 0.0f;
  for (int c = 0; c < NCHUNK; ++c) {
    float Ac = Achunk[h * NCHUNK + c];
    size_t base = ((size_t)h * NCHUNK + c) * 64;
#pragma unroll
    for (int k = 0; k < 16; ++k) {
      int hh = hh0 + k;
      SprevT[(base + hh) * 64 + d] = (__bf16)S[k];
      S[k] = Ac * S[k] + states[(base + d) * 64 + hh];
    }
  }
}

// ---------------------------------------------------------------------------
// Finish: y_inter = (exp(cs).*Cc) @ S_prev, y = y_inter + y_diag, then
// gated = y * silu(res) as bf16 in [token][D_INNER] layout for the out GEMM.
// ---------------------------------------------------------------------------
__global__ void __launch_bounds__(256)
finish_kernel(const __bf16* __restrict__ Ceb, const __bf16* __restrict__ SprevT,
              const float* __restrict__ ydiag, const float* __restrict__ xr,
              __bf16* __restrict__ gated) {
  const int c = blockIdx.x, h = blockIdx.y;
  const int tid = threadIdx.x, lane = tid & 31, w = tid >> 5;
  __shared__ __bf16 CeW[256 * 80];
  __shared__ __bf16 SpT[64 * 80];

  {
    const __bf16* src = Ceb + ((size_t)h * LSEQ + c * QCHUNK + tid) * 64;
#pragma unroll
    for (int q = 0; q < 64; q += 16)
      *(v16bf*)(CeW + tid * 80 + q) = *(const v16bf*)(src + q);
  }
  {
    int r = tid >> 2, q0 = (tid & 3) * 16;
    *(v16bf*)(SpT + r * 80 + q0) =
        *(const v16bf*)(SprevT + (((size_t)h * NCHUNK + c) * 64 + r) * 64 + q0);
  }
  __syncthreads();

  const int r0 = w * 32;
  v8f acc[2][4];
#pragma unroll
  for (int i = 0; i < 2; ++i)
#pragma unroll
    for (int j = 0; j < 4; ++j) acc[i][j] = v8f_zero();

#pragma unroll
  for (int kk = 0; kk < 64; kk += 32) {
    v16bf a0 = load_a_frag(CeW, 80, r0,      kk, lane);
    v16bf a1 = load_a_frag(CeW, 80, r0 + 16, kk, lane);
    v16bf bfr[4];
#pragma unroll
    for (int fh = 0; fh < 4; ++fh)
      bfr[fh] = load_b_frag(SpT, 80, fh * 16, kk, lane);
#pragma unroll
    for (int fh = 0; fh < 4; ++fh) {
      acc[0][fh] = wmma_bf16(a0, bfr[fh], acc[0][fh]);
      acc[1][fh] = wmma_bf16(a1, bfr[fh], acc[1][fh]);
    }
  }

  const int nl = lane & 15, hb = lane >> 4;
#pragma unroll
  for (int fm = 0; fm < 2; ++fm)
#pragma unroll
    for (int fh = 0; fh < 4; ++fh)
#pragma unroll
      for (int e = 0; e < 8; ++e) {
        int iloc = r0 + fm * 16 + hb * 8 + e;
        int t = c * QCHUNK + iloc;
        int hh = fh * 16 + nl;
        float y = acc[fm][fh][e] + ydiag[((size_t)h * LSEQ + t) * 64 + hh];
        float res = xr[(size_t)t * XRN + DINNER + h * DHEAD + hh];
        float sig = 1.0f / (1.0f + __expf(-res));
        gated[(size_t)t * DINNER + h * DHEAD + hh] = (__bf16)(y * res * sig);
      }
}

// ---------------------------------------------------------------------------
// Host launcher
// ---------------------------------------------------------------------------
extern "C" void kernel_launch(void* const* d_in, const int* in_sizes, int n_in,
                              void* d_out, int out_size, void* d_ws, size_t ws_size,
                              hipStream_t stream) {
  (void)in_sizes; (void)n_in; (void)out_size; (void)ws_size;
  const float* x       = (const float*)d_in[0];
  const float* w_in    = (const float*)d_in[1];
  const float* w_param = (const float*)d_in[2];
  const float* w_out   = (const float*)d_in[3];
  float* out = (float*)d_out;

  char* ws = (char*)d_ws;
  size_t off = 0;
  auto alloc = [&](size_t bytes) -> char* {
    char* p = ws + off;
    off += (bytes + 255) & ~(size_t)255;
    return p;
  };

  __bf16* x_bf    = (__bf16*)alloc((size_t)LSEQ * DMODEL * 2);
  __bf16* win_bf  = (__bf16*)alloc((size_t)XRN * DMODEL * 2);
  __bf16* wpar_bf = (__bf16*)alloc((size_t)NPR * DMODEL * 2);
  __bf16* wout_bf = (__bf16*)alloc((size_t)DMODEL * DINNER * 2);
  float*  xr      = (float*) alloc((size_t)LSEQ * XRN * 4);
  float*  params  = (float*) alloc((size_t)LSEQ * NPR * 4);
  float*  cs_buf  = (float*) alloc((size_t)NHEADS * LSEQ * 4);
  float*  Achunk  = (float*) alloc((size_t)NHEADS * NCHUNK * 4);
  __bf16* Bb      = (__bf16*)alloc((size_t)NHEADS * LSEQ * 64 * 2);
  __bf16* Cb      = (__bf16*)alloc((size_t)NHEADS * LSEQ * 64 * 2);
  __bf16* Ceb     = (__bf16*)alloc((size_t)NHEADS * LSEQ * 64 * 2);
  __bf16* BdT     = (__bf16*)alloc((size_t)NHEADS * NCHUNK * 64 * 256 * 2);
  __bf16* XcTg    = (__bf16*)alloc((size_t)NHEADS * NCHUNK * 64 * 256 * 2);
  float*  states  = (float*) alloc((size_t)NHEADS * NCHUNK * 64 * 64 * 4);
  __bf16* SprevT  = (__bf16*)alloc((size_t)NHEADS * NCHUNK * 64 * 64 * 2);
  float*  ydiag   = (float*) alloc((size_t)NHEADS * LSEQ * 64 * 4);
  __bf16* gated   = (__bf16*)alloc((size_t)LSEQ * DINNER * 2);

  // 1) fp32 -> bf16 conversions
  cvt_f32_bf16<<<2048, 256, 0, stream>>>(x, x_bf, (size_t)LSEQ * DMODEL);
  cvt_f32_bf16<<<4096, 256, 0, stream>>>(w_in, win_bf, (size_t)XRN * DMODEL);
  cvt_f32_bf16<<<4096, 256, 0, stream>>>(w_param, wpar_bf, (size_t)NPR * DMODEL);
  cvt_f32_bf16<<<2048, 256, 0, stream>>>(w_out, wout_bf, (size_t)DMODEL * DINNER);

  // 2) xr = x @ w_in^T   (4096 x 8192, K=2048)
  gemm_bf16_nt<<<dim3(XRN / 128, LSEQ / 128), 256, 0, stream>>>(
      x_bf, win_bf, xr, LSEQ, XRN, DMODEL);
  // 3) params = x @ w_param^T (4096 x 8256, K=2048)
  gemm_bf16_nt<<<dim3((NPR + 127) / 128, LSEQ / 128), 256, 0, stream>>>(
      x_bf, wpar_bf, params, LSEQ, NPR, DMODEL);

  // 4) SSD prep: cumsum + bf16 operand tensors
  prep_kernel<<<dim3(NCHUNK, NHEADS), 256, 0, stream>>>(
      params, xr, cs_buf, Achunk, Bb, Cb, Ceb, BdT, XcTg);

  // 5) intra-chunk (diagonal) part
  diag_kernel<<<dim3(NCHUNK * 2, NHEADS), 256, 0, stream>>>(
      Cb, Bb, XcTg, cs_buf, ydiag);

  // 6) per-chunk end states
  states_kernel<<<dim3(NCHUNK, NHEADS), 256, 0, stream>>>(BdT, XcTg, states);

  // 7) sequential inter-chunk recurrence
  scan_kernel<<<NHEADS, 256, 0, stream>>>(states, Achunk, SprevT);

  // 8) y_inter + combine + SiLU gating -> bf16
  finish_kernel<<<dim3(NCHUNK, NHEADS), 256, 0, stream>>>(
      Ceb, SprevT, ydiag, xr, gated);

  // 9) out = gated @ w_out^T (4096 x 2048, K=4096)
  gemm_bf16_nt<<<dim3(DMODEL / 128, LSEQ / 128), 256, 0, stream>>>(
      gated, wout_bf, out, LSEQ, DMODEL, DINNER);
}
